// AdvancedMoleculeGNN_45320494907638
// MI455X (gfx1250) — compile-verified, matrix operand order checked by reading, not run on hardware
//
#include <hip/hip_runtime.h>
#include <hip/hip_bf16.h>

#define N_NODES  200000
#define N_EDGES  600000
#define N_GRAPHS 10000
#define F_NODE   32
#define HDIM     128
#define BN_EPS   1e-5f

typedef __attribute__((ext_vector_type(16))) _Float16 v16h;
typedef __attribute__((ext_vector_type(8)))  _Float16 v8h;
typedef __attribute__((ext_vector_type(8)))  float    v8f;
typedef __attribute__((ext_vector_type(4)))  unsigned int u32x4;
typedef __attribute__((ext_vector_type(8)))  int      i32x8;
typedef __attribute__((ext_vector_type(4)))  int      i32x4;

// ---------------------------------------------------------------------------
// Prep: convert w1/w2 to f16 in WMMA B-fragment-swizzled order, fold BN.
// B-fragment layout (16-bit B, 32x16): lane L -> col n = L&15, khalf = L>>4;
// element j<8  -> k = kt*32 + khalf*8 + j
// element j>=8 -> k = kt*32 + 16 + khalf*8 + (j-8)
// Stored as wS[((nt*KT + kt)*32 + L)*16 + j] so a fragment is one 32B load.
// Also: bn_a = gamma*rsqrt(rvar+eps); bn_c = (b1 - rmean)*bn_a + beta.
// ---------------------------------------------------------------------------
__global__ void prep_kernel(const float* __restrict__ w1, const float* __restrict__ b1,
                            const float* __restrict__ gamma, const float* __restrict__ beta,
                            const float* __restrict__ rmean, const float* __restrict__ rvar,
                            const float* __restrict__ w2,
                            _Float16* __restrict__ w1s, _Float16* __restrict__ w2s,
                            float* __restrict__ bn_a, float* __restrict__ bn_c, int din)
{
    const int idx = blockIdx.x * blockDim.x + threadIdx.x;
    const int KT1 = din / 32;
    if (idx < 8 * KT1 * 512) {                       // w1: din x 128
        int j  = idx & 15;
        int L  = (idx >> 4) & 31;
        int fi = idx >> 9;
        int kt = fi % KT1, nt = fi / KT1;
        int n = L & 15, kh = L >> 4;
        int k = kt * 32 + kh * 8 + (j < 8 ? j : 16 + (j - 8));
        w1s[idx] = (_Float16)w1[k * HDIM + nt * 16 + n];
    }
    if (idx < 8 * 4 * 512) {                         // w2: 128 x 128
        int j  = idx & 15;
        int L  = (idx >> 4) & 31;
        int fi = idx >> 9;
        int kt = fi & 3, nt = fi >> 2;
        int n = L & 15, kh = L >> 4;
        int k = kt * 32 + kh * 8 + (j < 8 ? j : 16 + (j - 8));
        w2s[idx] = (_Float16)w2[k * HDIM + nt * 16 + n];
    }
    if (idx < HDIM) {
        float a = gamma[idx] * rsqrtf(rvar[idx] + BN_EPS);
        bn_a[idx] = a;
        bn_c[idx] = (b1[idx] - rmean[idx]) * a + beta[idx];
    }
}

// ---------------------------------------------------------------------------
// Simple data movers
// ---------------------------------------------------------------------------
__global__ void copy_f32(const float* __restrict__ in, float* __restrict__ out, long n) {
    long i = (long)blockIdx.x * blockDim.x + threadIdx.x;
    if (i < n) out[i] = in[i];
}
__global__ void zero_f32(float* __restrict__ out, long n) {
    long i = (long)blockIdx.x * blockDim.x + threadIdx.x;
    if (i < n) out[i] = 0.0f;
}

// Edge scatter-add: agg[dst] += feat[src]   (agg pre-initialized with feat -> h = x + sum)
template<int F>
__global__ void scatter_add(const float* __restrict__ feat, const int* __restrict__ esrc,
                            const int* __restrict__ edst, float* __restrict__ agg) {
    long i = (long)blockIdx.x * blockDim.x + threadIdx.x;
    if (i >= (long)N_EDGES * F) return;
    int e = (int)(i / F), f = (int)(i % F);
    atomicAdd(&agg[(size_t)edst[e] * F + f], feat[(size_t)esrc[e] * F + f]);
}

// Global add pool: pooled[batch[node]] += h[node]
__global__ void pool_add(const float* __restrict__ h, const int* __restrict__ batch,
                         float* __restrict__ pooled) {
    long i = (long)blockIdx.x * blockDim.x + threadIdx.x;
    if (i >= (long)N_NODES * HDIM) return;
    int node = (int)(i / HDIM), f = (int)(i % HDIM);
    atomicAdd(&pooled[(size_t)batch[node] * HDIM + f], h[i]);
}

// ---------------------------------------------------------------------------
// TDM helper: DMA `bytes` (multiple of 4, <= 256KB) from global into LDS as a
// 1-row 2D tile of DWORDs.  D# per CDNA5 ISA:
//   group0: [1:0]=count=1, [63:32]=lds_addr, [120:64]=global_addr, type=2
//   group1: data_size=4B (code 2); tensor_dim0 = tile_dim0 = ndwords;
//           tensor_dim1 = tile_dim1 = 1; tensor_dim0_stride = ndwords
//   groups 2/3: zero (tensor rank <= 2)
// ---------------------------------------------------------------------------
__device__ __forceinline__ void tdm_load_to_lds(void* lds_dst, const void* global_src,
                                                unsigned bytes)
{
    const unsigned nd = bytes >> 2;                       // DWORD count
    const unsigned lds_off = (unsigned)(size_t)lds_dst;   // flat LDS addr low 32b = offset
    const unsigned long long ga = (unsigned long long)(size_t)global_src;
    u32x4 g0 = { 1u,                                      // count=1, user descriptor
                 lds_off,
                 (unsigned)ga,
                 (unsigned)((ga >> 32) & 0x01FFFFFFu) | (2u << 30) };  // type=2 (image)
    i32x8 g1 = { (int)0x00020000u,                        // data_size = 2 (4 bytes)
                 (int)((nd & 0xFFFFu) << 16),             // tensor_dim0[15:0]
                 (int)((nd >> 16) | (1u << 16)),          // tensor_dim0[31:16], tensor_dim1=1
                 (int)((nd & 0xFFFFu) << 16),             // tile_dim0 = nd
                 1,                                       // tile_dim1 = 1
                 (int)nd,                                 // tensor_dim0_stride[31:0]
                 0, 0 };
    i32x4 gz = { 0, 0, 0, 0 };
#if defined(__clang_major__) && (__clang_major__ >= 23)
    i32x8 gz8 = { 0, 0, 0, 0, 0, 0, 0, 0 };
    __builtin_amdgcn_tensor_load_to_lds(g0, g1, gz, gz, gz8, 0);
#else
    __builtin_amdgcn_tensor_load_to_lds(g0, g1, gz, gz, 0);
#endif
}

// ---------------------------------------------------------------------------
// Fused GIN MLP on WMMA: out = relu( relu(bn(agg @ w1 + b1)) @ w2 + b2 )
// 128 threads = 4 waves; each wave owns one 16-row node tile.
// 200000 rows = 3125 blocks * 4 waves * 16 rows exactly -> EXEC all-ones.
// w2 (32 KB, shared by all 4 waves) is DMA'd into LDS once per block via TDM.
// ---------------------------------------------------------------------------
template<int DIN>
__global__ __launch_bounds__(128) void gin_mlp(
    const float* __restrict__ agg, const _Float16* __restrict__ w1s,
    const float* __restrict__ bn_a, const float* __restrict__ bn_c,
    const _Float16* __restrict__ w2s, const float* __restrict__ b2,
    float* __restrict__ out)
{
    constexpr int KT1 = DIN / 32;
    __shared__ alignas(16) _Float16 ldsA[4][16 * DIN];
    __shared__ alignas(16) _Float16 ldsT[4][16 * HDIM];
    __shared__ alignas(16) _Float16 ldsW2[HDIM * HDIM];   // block-shared w2 fragments

    const int lane = threadIdx.x & 31;
    const int wv   = threadIdx.x >> 5;
    const int tile = blockIdx.x * 4 + wv;
    const size_t row0 = (size_t)tile * 16;

    _Float16* A = ldsA[wv];
    _Float16* T = ldsT[wv];

    // Wave 0: async-DMA the pre-swizzled w2 matrix into LDS (TENSORcnt path)
    if (wv == 0) {
        tdm_load_to_lds(&ldsW2[0], w2s, HDIM * HDIM * 2);
    }

    // All waves: stage 16 x DIN agg tile into LDS as f16 (row-major)
    for (int i = lane; i < 16 * DIN; i += 32)
        A[i] = (_Float16)agg[row0 * DIN + i];

    if (wv == 0) __builtin_amdgcn_s_wait_tensorcnt(0);
    __syncthreads();   // publishes TDM-written w2 + per-wave A tiles

    const int n  = lane & 15;   // also the A-matrix row (M) for this lane
    const int kh = lane >> 4;   // K-half select (ISA 16-bit A/B layout)
    const v16h* W1 = (const v16h*)w1s;
    const v16h* W2 = (const v16h*)ldsW2;

    // A fragments for matmul 1: VGPR0-3 = K[kh*8 .. kh*8+7], VGPR4-7 = +16
    v16h afrag[KT1];
#pragma unroll
    for (int kt = 0; kt < KT1; ++kt) {
        const int kb = kt * 32 + kh * 8;
        v8h lo = *(const v8h*)&A[n * DIN + kb];
        v8h hi = *(const v8h*)&A[n * DIN + kb + 16];
        afrag[kt] = __builtin_shufflevector(lo, hi, 0,1,2,3,4,5,6,7,8,9,10,11,12,13,14,15);
    }

    // Matmul 1 + fused BN + ReLU -> LDS intermediate (f16)
#pragma unroll
    for (int nt = 0; nt < 8; ++nt) {
        v8f acc = {};
#pragma unroll
        for (int kt = 0; kt < KT1; ++kt)
            acc = __builtin_amdgcn_wmma_f32_16x16x32_f16(
                false, afrag[kt], false, W1[(nt * KT1 + kt) * 32 + lane],
                (short)0, acc, false, false);
        const int ncol = nt * 16 + n;
        const float as = bn_a[ncol], cs = bn_c[ncol];
#pragma unroll
        for (int r = 0; r < 8; ++r) {        // C/D layout: m = r + kh*8, n = lane&15
            float v = acc[r] * as + cs;
            v = v > 0.0f ? v : 0.0f;
            T[(r + kh * 8) * HDIM + ncol] = (_Float16)v;
        }
    }
    __syncthreads();

    // A fragments for matmul 2 (K = 128)
    v16h tfrag[4];
#pragma unroll
    for (int kt = 0; kt < 4; ++kt) {
        const int kb = kt * 32 + kh * 8;
        v8h lo = *(const v8h*)&T[n * HDIM + kb];
        v8h hi = *(const v8h*)&T[n * HDIM + kb + 16];
        tfrag[kt] = __builtin_shufflevector(lo, hi, 0,1,2,3,4,5,6,7,8,9,10,11,12,13,14,15);
    }

    // Matmul 2 (weights from LDS) + bias + outer ReLU -> f32 output
#pragma unroll
    for (int nt = 0; nt < 8; ++nt) {
        v8f acc = {};
#pragma unroll
        for (int kt = 0; kt < 4; ++kt)
            acc = __builtin_amdgcn_wmma_f32_16x16x32_f16(
                false, tfrag[kt], false, W2[(nt * 4 + kt) * 32 + lane],
                (short)0, acc, false, false);
        const int ncol = nt * 16 + n;
        const float bb = b2[ncol];
#pragma unroll
        for (int r = 0; r < 8; ++r) {
            float v = acc[r] + bb;
            v = v > 0.0f ? v : 0.0f;
            out[(row0 + r + kh * 8) * HDIM + ncol] = v;
        }
    }
}

// ---------------------------------------------------------------------------
// FC head (tiny: 0.16 GFLOP total) — scalar kernels
// ---------------------------------------------------------------------------
__global__ void fc1_kernel(const float* __restrict__ pooled, const float* __restrict__ w,
                           const float* __restrict__ b, float* __restrict__ out) {
    int i = blockIdx.x * blockDim.x + threadIdx.x;
    if (i >= N_GRAPHS * 64) return;
    int g = i / 64, j = i % 64;
    float s = b[j];
    const float* pr = &pooled[(size_t)g * HDIM];
    for (int k = 0; k < HDIM; ++k) s += pr[k] * w[k * 64 + j];
    out[i] = s > 0.0f ? s : 0.0f;
}
__global__ void fc2_kernel(const float* __restrict__ x, const float* __restrict__ w,
                           const float* __restrict__ b, float* __restrict__ out) {
    int g = blockIdx.x * blockDim.x + threadIdx.x;
    if (g >= N_GRAPHS) return;
    float s = b[0];
    const float* xr = &x[(size_t)g * 64];
    for (int j = 0; j < 64; ++j) s += xr[j] * w[j];
    out[g] = s;
}

// ---------------------------------------------------------------------------
extern "C" void kernel_launch(void* const* d_in, const int* in_sizes, int n_in,
                              void* d_out, int out_size, void* d_ws, size_t ws_size,
                              hipStream_t stream) {
    (void)in_sizes; (void)n_in; (void)out_size; (void)ws_size;

    const float* x     = (const float*)d_in[0];
    const int*   eidx  = (const int*)d_in[1];
    const int*   batch = (const int*)d_in[3];
    const float *w1[3], *b1[3], *ga[3], *be[3], *rm[3], *rv[3], *w2[3], *b2[3];
    for (int l = 0; l < 3; ++l) {
        int base = 4 + 8 * l;
        w1[l] = (const float*)d_in[base + 0];
        b1[l] = (const float*)d_in[base + 1];
        ga[l] = (const float*)d_in[base + 2];
        be[l] = (const float*)d_in[base + 3];
        rm[l] = (const float*)d_in[base + 4];
        rv[l] = (const float*)d_in[base + 5];
        w2[l] = (const float*)d_in[base + 6];
        b2[l] = (const float*)d_in[base + 7];
    }
    const float* w_fc1 = (const float*)d_in[30];
    const float* b_fc1 = (const float*)d_in[31];
    const float* w_fc2 = (const float*)d_in[32];
    const float* b_fc2 = (const float*)d_in[33];

    char* ws = (char*)d_ws;
    size_t off = 0;
    auto alloc = [&](size_t bytes) -> char* {
        char* p = ws + off;
        off += (bytes + 255) & ~(size_t)255;
        return p;
    };
    float* bufA   = (float*)alloc((size_t)N_NODES * HDIM * 4);   // current node features
    float* bufB   = (float*)alloc((size_t)N_NODES * HDIM * 4);   // aggregation buffer
    float* pooled = (float*)alloc((size_t)N_GRAPHS * HDIM * 4);
    float* fc1o   = (float*)alloc((size_t)N_GRAPHS * 64 * 4);
    _Float16* w1s[3]; _Float16* w2s[3]; float* bna[3]; float* bnc[3];
    for (int l = 0; l < 3; ++l) {
        int din = (l == 0) ? F_NODE : HDIM;
        w1s[l] = (_Float16*)alloc((size_t)din * HDIM * 2);
        w2s[l] = (_Float16*)alloc((size_t)HDIM * HDIM * 2);
        bna[l] = (float*)alloc(HDIM * 4);
        bnc[l] = (float*)alloc(HDIM * 4);
    }
    const int* esrc = eidx;
    const int* edst = eidx + N_EDGES;

    // Weight prep (covers max 8*4*512 = 16384 elements -> 64 blocks)
    for (int l = 0; l < 3; ++l) {
        int din = (l == 0) ? F_NODE : HDIM;
        prep_kernel<<<64, 256, 0, stream>>>(w1[l], b1[l], ga[l], be[l], rm[l], rv[l], w2[l],
                                            w1s[l], w2s[l], bna[l], bnc[l], din);
    }

    // Layer 1 (din = 32)
    {
        long n = (long)N_NODES * F_NODE;
        copy_f32<<<(unsigned)((n + 255) / 256), 256, 0, stream>>>(x, bufB, n);
        long t = (long)N_EDGES * F_NODE;
        scatter_add<F_NODE><<<(unsigned)((t + 255) / 256), 256, 0, stream>>>(x, esrc, edst, bufB);
        gin_mlp<F_NODE><<<N_NODES / 64, 128, 0, stream>>>(bufB, w1s[0], bna[0], bnc[0],
                                                          w2s[0], b2[0], bufA);
    }
    // Layers 2, 3 (din = 128)
    for (int l = 1; l < 3; ++l) {
        long n = (long)N_NODES * HDIM;
        copy_f32<<<(unsigned)((n + 255) / 256), 256, 0, stream>>>(bufA, bufB, n);
        long t = (long)N_EDGES * HDIM;
        scatter_add<HDIM><<<(unsigned)((t + 255) / 256), 256, 0, stream>>>(bufA, esrc, edst, bufB);
        gin_mlp<HDIM><<<N_NODES / 64, 128, 0, stream>>>(bufB, w1s[l], bna[l], bnc[l],
                                                        w2s[l], b2[l], bufA);
    }

    // Pool + FC head
    zero_f32<<<(N_GRAPHS * HDIM + 255) / 256, 256, 0, stream>>>(pooled, (long)N_GRAPHS * HDIM);
    pool_add<<<(unsigned)(((long)N_NODES * HDIM + 255) / 256), 256, 0, stream>>>(bufA, batch, pooled);
    fc1_kernel<<<(N_GRAPHS * 64 + 255) / 256, 256, 0, stream>>>(pooled, w_fc1, b_fc1, fc1o);
    fc2_kernel<<<(N_GRAPHS + 255) / 256, 256, 0, stream>>>(fc1o, w_fc2, b_fc2, (float*)d_out);
}
